// RPGCN_75050258530750
// MI455X (gfx1250) — matrix operand order repaired and checked by reading, not compiled, not run on hardware
//
#include <hip/hip_runtime.h>

// ---------------------------------------------------------------------------
// RPGCN for MI455X (gfx1250): bf16 WMMA GEMMs + TDM weight staging +
// L2-resident atomic scatter.
//   u = (h @ W) * dinv          (v_wmma_f32_16x16x32_bf16, dinv folded in)
//   acc[d] += u[s]              (edge scatter, global_atomic_add_f32, L2)
//   h' = relu(dinv*(acc+u) + b) (finalize; +u is self loop; emits packed bf16)
// Activations live in global memory as packed bf16 pairs so each WMMA A
// fragment is exactly two global_load_b128s (no per-step cvt VALU).
// ---------------------------------------------------------------------------

#define NN     100000
#define EE     1600000
#define GG     500
#define NPGc   200
#define XSc    128
#define FIXc   10
#define NPERMc 3
#define MT     (NN / 16)   // 6250 16-row M tiles (N divisible by 16)

typedef __bf16 bf16_t;
typedef bf16_t v16bf __attribute__((ext_vector_type(16)));
typedef float  v8f   __attribute__((ext_vector_type(8)));
typedef unsigned int u32x4 __attribute__((ext_vector_type(4)));
typedef int i32x8 __attribute__((ext_vector_type(8)));
typedef int i32x4 __attribute__((ext_vector_type(4)));

union Frag16 { v16bf v; unsigned u[8]; };

__device__ __forceinline__ unsigned pack_bf2(float lo, float hi) {
  unsigned a = __float_as_uint(lo), b = __float_as_uint(hi);
  a = (a + 0x7FFFu + ((a >> 16) & 1u)) >> 16;   // RNE f32 -> bf16
  b = (b + 0x7FFFu + ((b >> 16) & 1u)) >> 16;
  return (a & 0xFFFFu) | (b << 16);
}
__device__ __forceinline__ unsigned short f2bf(float f) {
  unsigned a = __float_as_uint(f);
  return (unsigned short)((a + 0x7FFFu + ((a >> 16) & 1u)) >> 16);
}

// ---------------------------------------------------------------------------
// Setup kernels
// ---------------------------------------------------------------------------
__global__ __launch_bounds__(256) void init_k(int* __restrict__ deg,
                                              float* __restrict__ acc,
                                              float* __restrict__ outg) {
  size_t tid = (size_t)blockIdx.x * 256 + threadIdx.x;
  if (tid < (size_t)NN * 64) acc[tid] = 0.0f;
  if (tid < NN)              deg[tid] = 1;          // self loop
  if (tid < (size_t)GG * 64) outg[tid] = 0.0f;
}

__global__ __launch_bounds__(256) void deg_k(const int* __restrict__ dst,
                                             int* __restrict__ deg) {
  int e = blockIdx.x * 256 + threadIdx.x;
  if (e < EE) atomicAdd(&deg[dst[e]], 1);
}

__global__ __launch_bounds__(256) void dinv_k(const int* __restrict__ deg,
                                              float* __restrict__ dinv) {
  int n = blockIdx.x * 256 + threadIdx.x;
  if (n < NN) dinv[n] = rsqrtf((float)deg[n]);
}

// f32 pairs -> packed bf16 dwords (used once for x).
__global__ __launch_bounds__(256) void cvt_pack_k(const float* __restrict__ s,
                                                  unsigned* __restrict__ d,
                                                  long long npairs) {
  long long t = (long long)blockIdx.x * 256 + threadIdx.x;
  if (t < npairs) d[t] = pack_bf2(s[2 * t], s[2 * t + 1]);
}

// ---------------------------------------------------------------------------
// GEMM: U[N,64] = (bf16(A)[N,2*LDAU] @ W[KMAIN,64] (+ pid @ Wx)) * dinv[row]
// One wave owns a 16x64 output strip (4 WMMA accumulators).
// Weights are DMA'd global->LDS by the Tensor Data Mover, then converted to
// a transposed bf16 [64][KMAIN] layout so each B-fragment VGPR is one aligned
// 32-bit LDS read (a K-pair); the compiler merges those into ds_load_b128.
// A is packed bf16 in global memory: 2 x global_load_b128 per k-step.
// ---------------------------------------------------------------------------
template <int KMAIN, int LDAU, bool HAS_PID>
__global__ __launch_bounds__(256) void gemm_conv(
    const unsigned* __restrict__ A, const float* __restrict__ W,
    const float* __restrict__ Wx, const float* __restrict__ pid,
    const float* __restrict__ dinv, float* __restrict__ U) {
  __shared__ __align__(16) float lds_raw[KMAIN * 64];
  __shared__ __align__(16) unsigned short wT[64 * KMAIN];

  // --- TDM: DMA the f32 weight tile [KMAIN,64] into LDS (wave 0 issues) ---
  if (threadIdx.x < 32) {
    unsigned long long ga = (unsigned long long)(uintptr_t)W;
    u32x4 g0;
    g0[0] = 1u;                                        // count=1 (valid D#)
    g0[1] = (unsigned)(uintptr_t)(&lds_raw[0]);        // lds_addr (bytes)
    g0[2] = (unsigned)ga;                              // global_addr[31:0]
    g0[3] = (unsigned)((ga >> 32) & 0x1FFFFFFull)      // global_addr[56:32]
            | (2u << 30);                              // type=2 ("image")
    i32x8 g1;
    g1[0] = 0x00020000;                                // data_size=2 -> 4B
    g1[1] = 64 << 16;                                  // tensor_dim0 = 64
    g1[2] = KMAIN << 16;                               // tensor_dim1 = KMAIN
    g1[3] = 64 << 16;                                  // tile_dim0 = 64
    g1[4] = KMAIN;                                     // tile_dim1 = KMAIN
    g1[5] = 64;                                        // tensor_dim0_stride
    g1[6] = 0;
    g1[7] = 0;
    i32x4 gz4 = {0, 0, 0, 0};                          // dims 2-4 unused
    i32x8 gz8 = {0, 0, 0, 0, 0, 0, 0, 0};
    __builtin_amdgcn_tensor_load_to_lds(g0, g1, gz4, gz4, gz8, 0);
    __builtin_amdgcn_s_wait_tensorcnt(0);
  }
  __syncthreads();

  // LDS f32 -> transposed bf16 [n][k]
  for (int idx = threadIdx.x; idx < 64 * KMAIN; idx += 256) {
    int n = idx & 63, k = idx >> 6;
    wT[n * KMAIN + k] = f2bf(lds_raw[idx]);
  }
  __syncthreads();

  int wave = threadIdx.x >> 5, lane = threadIdx.x & 31;
  int tile = blockIdx.x * 8 + wave;
  if (tile >= MT) return;

  int row_a = tile * 16 + (lane & 15);                // A-fragment row
  const uint4* ap = (const uint4*)(A + (size_t)row_a * LDAU);

  v8f acc4[4] = {};                                   // 16x64 f32 strip

  for (int kt = 0; kt < KMAIN / 32; ++kt) {
    // A: K pairs {kb..kb+7} and {kb+16..kb+23}, kb = kt*32 + (lane<16?0:8)
    int q = kt * 4 + ((lane < 16) ? 0 : 1);
    uint4 lo = ap[q];
    uint4 hi = ap[q + 2];
    Frag16 a;
    a.u[0] = lo.x; a.u[1] = lo.y; a.u[2] = lo.z; a.u[3] = lo.w;
    a.u[4] = hi.x; a.u[5] = hi.y; a.u[6] = hi.z; a.u[7] = hi.w;

    int ncol = lane & 15;
    int kb2  = kt * 32 + ((lane < 16) ? 0 : 8);       // B half-lane split
#pragma unroll
    for (int nt = 0; nt < 4; ++nt) {
      Frag16 b;
      const unsigned* wp =
          (const unsigned*)(wT + (nt * 16 + ncol) * KMAIN + kb2);
#pragma unroll
      for (int v = 0; v < 8; ++v) b.u[v] = wp[v < 4 ? v : v + 4];     // K-pairs per VGPR
      acc4[nt] = __builtin_amdgcn_wmma_f32_16x16x32_bf16(
          false, a.v, false, b.v, (short)0, acc4[nt], false, false);
    }
  }

  // Epilogue: + one-hot tail (layer 1), * dinv, store f32 u.
  int row_c = tile * 16 + ((lane & 16) ? 8 : 0);      // C rows 0-7 / 8-15
  int ncol  = lane & 15;
#pragma unroll
  for (int v = 0; v < 8; ++v) {
    int r = row_c + v;
    float scale = dinv[r];
    float extra[4] = {0.f, 0.f, 0.f, 0.f};
    if (HAS_PID) {
      const float* pr = pid + (size_t)r * FIXc;
#pragma unroll
      for (int j = 0; j < FIXc; ++j) {
        float pj = pr[j];
#pragma unroll
        for (int nt = 0; nt < 4; ++nt)
          extra[nt] += pj * Wx[j * 64 + nt * 16 + ncol];
      }
    }
#pragma unroll
    for (int nt = 0; nt < 4; ++nt)
      U[(size_t)r * 64 + nt * 16 + ncol] = (acc4[nt][v] + extra[nt]) * scale;
  }
}

// ---------------------------------------------------------------------------
// Edge scatter: acc[dst] += u[src]. 4 threads/edge, float4 gathers, f32
// atomics (global_atomic_add_f32) — all L2-resident (u+acc = 51 MB << 192 MB).
// ---------------------------------------------------------------------------
__global__ __launch_bounds__(256) void edge_scatter(
    const int* __restrict__ src, const int* __restrict__ dst,
    const float* __restrict__ u, float* __restrict__ acc) {
  long long tid = (long long)blockIdx.x * 256 + threadIdx.x;
  int e = (int)(tid >> 2);
  if (e >= EE) return;
  int c = ((int)tid & 3) << 4;                        // 16-float chunk
  int s = src[e], d = dst[e];
  const float4* us = (const float4*)(u + (size_t)s * 64 + c);
  float* ad = acc + (size_t)d * 64 + c;
#pragma unroll
  for (int i = 0; i < 4; ++i) {
    float4 vv = us[i];
    atomicAdd(ad + 4 * i + 0, vv.x);
    atomicAdd(ad + 4 * i + 1, vv.y);
    atomicAdd(ad + 4 * i + 2, vv.z);
    atomicAdd(ad + 4 * i + 3, vv.w);
  }
}

// h = relu(dinv*(acc + u) + b) emitted as packed bf16; acc reset to 0.
__global__ __launch_bounds__(256) void finalize_k(
    const float* __restrict__ u, float* __restrict__ acc,
    const float* __restrict__ b, const float* __restrict__ dinv,
    unsigned* __restrict__ hb) {
  size_t tid = (size_t)blockIdx.x * 256 + threadIdx.x;
  if (tid >= (size_t)NN * 16) return;
  int n = (int)(tid >> 4);
  int c = ((int)tid & 15) << 2;
  float di = dinv[n];
  size_t idx = (size_t)n * 64 + c;
  float4 a  = *(const float4*)(acc + idx);
  float4 uu = *(const float4*)(u + idx);
  float r0 = fmaxf(di * (a.x + uu.x) + b[c + 0], 0.f);
  float r1 = fmaxf(di * (a.y + uu.y) + b[c + 1], 0.f);
  float r2 = fmaxf(di * (a.z + uu.z) + b[c + 2], 0.f);
  float r3 = fmaxf(di * (a.w + uu.w) + b[c + 3], 0.f);
  *(uint2*)(hb + (size_t)n * 32 + (c >> 1)) =
      make_uint2(pack_bf2(r0, r1), pack_bf2(r2, r3));
  *(float4*)(acc + idx) = make_float4(0.f, 0.f, 0.f, 0.f);
}

// Deterministic per-graph mean pool (no atomics) over packed bf16 h.
__global__ __launch_bounds__(256) void pool_k(const unsigned* __restrict__ hb,
                                              float* __restrict__ outg) {
  int tid = blockIdx.x * 256 + threadIdx.x;
  if (tid >= GG * 64) return;
  int g = tid >> 6, c = tid & 63;
  const unsigned* p = hb + (size_t)g * NPGc * 32 + (c >> 1);
  bool hi = (c & 1) != 0;
  float s = 0.f;
  for (int i = 0; i < NPGc; ++i) {
    unsigned w = p[i * 32];
    s += __uint_as_float(hi ? (w & 0xFFFF0000u) : (w << 16));
  }
  outg[tid] += s * (1.0f / (float(NPGc) * float(NPERMc)));
}

// Tiny head: y[g] = relu(outg[g] @ a1w + a1b) @ a2w + a2b   (1 MFLOP total)
__global__ __launch_bounds__(256) void mlp_k(
    const float* __restrict__ outg, const float* __restrict__ a1w,
    const float* __restrict__ a1b, const float* __restrict__ a2w,
    const float* __restrict__ a2b, float* __restrict__ y) {
  int g = blockIdx.x * 256 + threadIdx.x;
  if (g >= GG) return;
  const float* p = outg + (size_t)g * 64;
  float acc2 = a2b[0];
  for (int j = 0; j < 16; ++j) {
    float t = a1b[j];
    for (int k = 0; k < 64; ++k) t += p[k] * a1w[k * 16 + j];
    acc2 += fmaxf(t, 0.f) * a2w[j];
  }
  y[g] = acc2;
}

// ---------------------------------------------------------------------------
extern "C" void kernel_launch(void* const* d_in, const int* in_sizes, int n_in,
                              void* d_out, int out_size, void* d_ws,
                              size_t ws_size, hipStream_t stream) {
  const float* x   = (const float*)d_in[0];
  const int*   ei  = (const int*)d_in[1];
  const float* pid = (const float*)d_in[3];   // [NPERM, N, FIX]
  const float* w1  = (const float*)d_in[4];   // [138, 64]
  const float* b1  = (const float*)d_in[5];
  const float* w2  = (const float*)d_in[6];   // [64, 64]
  const float* b2  = (const float*)d_in[7];
  const float* w3  = (const float*)d_in[8];
  const float* b3  = (const float*)d_in[9];
  const float* a1w = (const float*)d_in[10];  // [64, 16]
  const float* a1b = (const float*)d_in[11];
  const float* a2w = (const float*)d_in[12];  // [16, 1]
  const float* a2b = (const float*)d_in[13];
  float* y = (float*)d_out;
  (void)in_sizes; (void)n_in; (void)out_size; (void)ws_size;

  const int* src = ei;
  const int* dst = ei + EE;

  // Workspace layout (~91 MB total, all 256B-aligned).
  char* ws = (char*)d_ws;
  size_t off = 0;
  auto take = [&](size_t bytes) {
    void* p = ws + off;
    off = (off + bytes + 255) & ~(size_t)255;
    return p;
  };
  int*      deg  = (int*)     take((size_t)NN * 4);
  float*    dinv = (float*)   take((size_t)NN * 4);
  float*    u    = (float*)   take((size_t)NN * 64 * 4);
  float*    acc  = (float*)   take((size_t)NN * 64 * 4);
  unsigned* xb   = (unsigned*)take((size_t)NN * 64 * 4);  // bf16-packed x
  unsigned* hb   = (unsigned*)take((size_t)NN * 32 * 4);  // bf16-packed h
  float*    outg = (float*)   take((size_t)GG * 64 * 4);

  dim3 blk(256);
  init_k<<<((size_t)NN * 64 + 255) / 256, blk, 0, stream>>>(deg, acc, outg);
  deg_k<<<(EE + 255) / 256, blk, 0, stream>>>(dst, deg);
  dinv_k<<<(NN + 255) / 256, blk, 0, stream>>>(deg, dinv);
  cvt_pack_k<<<((size_t)NN * 64 + 255) / 256, blk, 0, stream>>>(
      x, xb, (long long)NN * 64);

  const int gemm_blocks    = (MT + 7) / 8;          // 8 waves/block
  const int scatter_blocks = (EE * 4 + 255) / 256;
  const int fin_blocks     = (NN * 16 + 255) / 256;

  for (int p = 0; p < NPERMc; ++p) {
    const float* pidp = pid + (size_t)p * NN * FIXc;
    // Layer 1: K=128 via WMMA + 10-wide one-hot tail in VALU.
    gemm_conv<128, 64, true><<<gemm_blocks, blk, 0, stream>>>(
        xb, w1, w1 + 128 * 64, pidp, dinv, u);
    edge_scatter<<<scatter_blocks, blk, 0, stream>>>(src, dst, u, acc);
    finalize_k<<<fin_blocks, blk, 0, stream>>>(u, acc, b1, dinv, hb);
    // Layer 2
    gemm_conv<64, 32, false><<<gemm_blocks, blk, 0, stream>>>(
        hb, w2, nullptr, nullptr, dinv, u);
    edge_scatter<<<scatter_blocks, blk, 0, stream>>>(src, dst, u, acc);
    finalize_k<<<fin_blocks, blk, 0, stream>>>(u, acc, b2, dinv, hb);
    // Layer 3
    gemm_conv<64, 32, false><<<gemm_blocks, blk, 0, stream>>>(
        hb, w3, nullptr, nullptr, dinv, u);
    edge_scatter<<<scatter_blocks, blk, 0, stream>>>(src, dst, u, acc);
    finalize_k<<<fin_blocks, blk, 0, stream>>>(u, acc, b3, dinv, hb);

    pool_k<<<(GG * 64 + 255) / 256, blk, 0, stream>>>(hb, outg);
  }

  mlp_k<<<(GG + 255) / 256, blk, 0, stream>>>(outg, a1w, a1b, a2w, a2b, y);
}
